// Pix2StructVisionAttention_5609227288921
// MI455X (gfx1250) — matrix operand order repaired and checked by reading, not compiled, not run on hardware
//
#include <hip/hip_runtime.h>
#include <hip/hip_bf16.h>
#include <float.h>

#define NHEADS 12
#define DKV    64
#define HID    768
#define BATCH  2
#define SEQ    2048
#define MTOT   (BATCH*SEQ)   /* 4096 */
#define NEGMIN (-3.4028234663852886e38f)

typedef __bf16 bf16_t;
typedef __attribute__((ext_vector_type(16))) __bf16 v16bf;
typedef __attribute__((ext_vector_type(8)))  float  v8f;
typedef __attribute__((ext_vector_type(4)))  unsigned int v4u;

__device__ __forceinline__ unsigned bfbits(float f) {
    unsigned u = __builtin_bit_cast(unsigned, f);
    u += 0x7FFFu + ((u >> 16) & 1u);               // round-to-nearest-even
    return u >> 16;
}
__device__ __forceinline__ bf16_t f2bf(float f) {
    unsigned short s = (unsigned short)bfbits(f);
    return __builtin_bit_cast(bf16_t, s);
}
__device__ __forceinline__ unsigned packbf(float lo, float hi) {
    return (bfbits(hi) << 16) | bfbits(lo);
}

union ABld { v16bf v; v4u q[2]; };

// A operand (16x32, M x K): lane L<16 -> row L, K 0..15; lane L+16 -> row L, K 16..31
__device__ __forceinline__ v16bf load_tile_a(const bf16_t* base, int stride, int lane) {
    int r = lane & 15, h = lane >> 4;
    const bf16_t* p = base + r * stride + h * 16;
    ABld t;
    t.q[0] = *(const v4u*)(p);
    t.q[1] = *(const v4u*)(p + 8);
    return t.v;
}

// B operand (32x16, K x N) sourced from a row-major [N x K] array:
// lane L<16 -> column N=L, K 0..15 ; lane L+16 -> column N=L, K 16..31 (2 per VGPR)
__device__ __forceinline__ v16bf load_tile_b(const bf16_t* base, int stride, int lane) {
    int r = lane & 15, h = lane >> 4;
    const bf16_t* p = base + r * stride + h * 16;
    ABld t;
    t.q[0] = *(const v4u*)(p);
    t.q[1] = *(const v4u*)(p + 8);
    return t.v;
}

__device__ __forceinline__ v8f wmma_bf(v16bf a, v16bf b, v8f c) {
    return __builtin_amdgcn_wmma_f32_16x16x32_bf16(false, a, false, b, (short)0, c, false, false);
}

// ---------------------------------------------------------------- fp32 -> bf16
__global__ void cvt_kernel(const float* __restrict__ in, bf16_t* __restrict__ out, int n) {
    int i = (blockIdx.x * 256 + threadIdx.x) * 4;
    if (i < n) {
        float4 f = *(const float4*)(in + i);
        out[i + 0] = f2bf(f.x);
        out[i + 1] = f2bf(f.y);
        out[i + 2] = f2bf(f.z);
        out[i + 3] = f2bf(f.w);
    }
}

// ------------------------------------------------- fused QKV projection GEMM
// C[M=4096, N=768] = Xb * W^T ;  z selects Wq/Wk/Wv and the store layout.
// Wave tile 32(M) x 64(N) = 2x4 WMMA tiles; block = 4 M-waves x 2 N-waves = 128x128.
// Q,K stored [B,H,S,64] bf16 ; V stored transposed [B,H,64,S] bf16.
__global__ __launch_bounds__(256) void qkv_kernel(
        const bf16_t* __restrict__ Xb,
        const bf16_t* __restrict__ Wqb, const bf16_t* __restrict__ Wkb,
        const bf16_t* __restrict__ Wvb,
        bf16_t* __restrict__ Qb, bf16_t* __restrict__ Kb, bf16_t* __restrict__ Vtb) {
    int lane = threadIdx.x & 31;
    int wave = threadIdx.x >> 5;
    int wm = wave & 3, wn = wave >> 2;
    int M0 = blockIdx.x * 128 + wm * 32;
    int N0 = blockIdx.y * 128 + wn * 64;        // N0 is head-aligned (multiple of 64)
    int z  = blockIdx.z;
    const bf16_t* W = (z == 0) ? Wqb : (z == 1 ? Wkb : Wvb);

    v8f acc[2][4] = {};
    for (int k0 = 0; k0 < HID; k0 += 32) {
        v16bf a0 = load_tile_a(Xb + (size_t)(M0 +  0) * HID + k0, HID, lane);
        v16bf a1 = load_tile_a(Xb + (size_t)(M0 + 16) * HID + k0, HID, lane);
        #pragma unroll
        for (int nt = 0; nt < 4; nt++) {
            v16bf bt = load_tile_b(W + (size_t)(N0 + nt * 16) * HID + k0, HID, lane);
            acc[0][nt] = wmma_bf(a0, bt, acc[0][nt]);
            acc[1][nt] = wmma_bf(a1, bt, acc[1][nt]);
        }
    }

    int r  = lane & 15;
    int mb = (lane >> 4) * 8;
    int b  = M0 >> 11;           // 128-row block tiles never straddle a batch boundary
    int s0 = M0 & (SEQ - 1);
    int head = N0 >> 6;

    if (z < 2) {
        bf16_t* out = (z == 0) ? Qb : Kb;
        #pragma unroll
        for (int mt = 0; mt < 2; mt++)
            #pragma unroll
            for (int nt = 0; nt < 4; nt++) {
                int d = nt * 16 + r;
                int srow = s0 + mt * 16 + mb;
                bf16_t* dst = out + (((size_t)(b * NHEADS + head) * SEQ + srow) * DKV + d);
                #pragma unroll
                for (int v = 0; v < 8; v++)
                    dst[(size_t)v * DKV] = f2bf(acc[mt][nt][v]);
            }
    } else {
        // V transposed: element (s,d) -> Vt[(b,h,d,s)]; 8 consecutive s per lane -> b128 store
        #pragma unroll
        for (int mt = 0; mt < 2; mt++)
            #pragma unroll
            for (int nt = 0; nt < 4; nt++) {
                int d = nt * 16 + r;
                int srow = s0 + mt * 16 + mb;
                union { bf16_t h8[8]; v4u q; } pk;
                #pragma unroll
                for (int v = 0; v < 8; v++) pk.h8[v] = f2bf(acc[mt][nt][v]);
                *(v4u*)(Vtb + ((size_t)(b * NHEADS + head) * DKV + d) * SEQ + srow) = pk.q;
            }
    }
}

// ------------------------------------------------------------ flash attention
// Transposed formulation: S^T = K * Q^T so that lane index == query.
// One wave = 32 query cols; block = 8 waves = 256 queries; grid = (S/256, B*H).
// O^T = V^T * P accumulated in registers; stores are packed b128 row-major O.
__global__ __launch_bounds__(256) void attn_kernel(
        const bf16_t* __restrict__ Qb, const bf16_t* __restrict__ Kb,
        const bf16_t* __restrict__ Vtb, const float* __restrict__ mask,
        bf16_t* __restrict__ Ob) {
    int lane = threadIdx.x & 31;
    int wave = threadIdx.x >> 5;
    int bh = blockIdx.y;
    int b = bh / NHEADS, head = bh % NHEADS;
    int q0 = blockIdx.x * 256 + wave * 32;
    int r = lane & 15;
    int half = lane >> 4;

    const bf16_t* Qp  = Qb  + ((size_t)bh * SEQ + q0) * DKV;
    const bf16_t* Kp0 = Kb  + (size_t)bh * SEQ * DKV;
    const bf16_t* Vp  = Vtb + (size_t)bh * DKV * SEQ;

    // Q as B operand (hoisted): [N=q x K=d] rows, 2 q-tiles x 2 d-steps
    v16bf Qopb[2][2];
    #pragma unroll
    for (int nt = 0; nt < 2; nt++)
        #pragma unroll
        for (int ks = 0; ks < 2; ks++)
            Qopb[nt][ks] = load_tile_b(Qp + (size_t)(nt * 16) * DKV + ks * 32, DKV, lane);

    // per-lane softmax state (lane == query); same value in both lane halves
    float biasl[2], mrun[2], lrun[2];
    #pragma unroll
    for (int nt = 0; nt < 2; nt++) {
        int q = q0 + nt * 16 + r;
        biasl[nt] = (mask[(size_t)b * SEQ + q] != 0.f) ? 0.f : NEGMIN;
        mrun[nt] = NEGMIN;
        lrun[nt] = 0.f;
    }
    v8f Oacc[4][2] = {};   // [d-tile][q-tile], O^T layout: lane = q, vgpr = d

    for (int kc = 0; kc < SEQ; kc += 32) {
        const bf16_t* Kp = Kp0 + (size_t)kc * DKV;

        if (kc + 32 < SEQ) {   // prefetch next K chunk + next V columns (gfx1250 global_prefetch_b8)
            __builtin_prefetch((const char*)(Kp + 32 * DKV) + lane * 128, 0, 0);
            __builtin_prefetch((const char*)(Vp + (size_t)lane * SEQ + kc + 32), 0, 0);
            __builtin_prefetch((const char*)(Vp + (size_t)(lane + 32) * SEQ + kc + 32), 0, 0);
        }

        // S^T[key, q] : A = K rows (2 key-tiles), B = Q rows
        v8f Sacc[2][2] = {};
        #pragma unroll
        for (int ks = 0; ks < 2; ks++) {
            v16bf ka0 = load_tile_a(Kp + 0 * DKV + ks * 32, DKV, lane);
            v16bf ka1 = load_tile_a(Kp + 16 * DKV + ks * 32, DKV, lane);
            Sacc[0][0] = wmma_bf(ka0, Qopb[0][ks], Sacc[0][0]);
            Sacc[0][1] = wmma_bf(ka0, Qopb[1][ks], Sacc[0][1]);
            Sacc[1][0] = wmma_bf(ka1, Qopb[0][ks], Sacc[1][0]);
            Sacc[1][1] = wmma_bf(ka1, Qopb[1][ks], Sacc[1][1]);
        }
        // lane holds keys {0..7, 16..23} (half 0) or {8..15, 24..31} (half 1) of this chunk
        // for queries q = q0 + nt*16 + r.
        #pragma unroll
        for (int nt = 0; nt < 2; nt++) {
            float sv[16], pv[16];
            #pragma unroll
            for (int j = 0; j < 8; j++) {
                sv[j]     = fmaxf(Sacc[0][nt][j] + biasl[nt], NEGMIN);
                sv[8 + j] = fmaxf(Sacc[1][nt][j] + biasl[nt], NEGMIN);
            }
            float lm = sv[0];
            #pragma unroll
            for (int j = 1; j < 16; j++) lm = fmaxf(lm, sv[j]);
            lm = fmaxf(lm, __shfl_xor(lm, 16, 32));          // combine lane halves
            float mnew = fmaxf(mrun[nt], lm);
            float ls = 0.f;
            #pragma unroll
            for (int j = 0; j < 16; j++) { pv[j] = __expf(sv[j] - mnew); ls += pv[j]; }
            ls += __shfl_xor(ls, 16, 32);
            float sc = __expf(mrun[nt] - mnew);
            lrun[nt] = lrun[nt] * sc + ls;
            mrun[nt] = mnew;
            // rescale O^T accumulators for this q-tile (scale is lane-resident)
            #pragma unroll
            for (int dt = 0; dt < 4; dt++)
                #pragma unroll
                for (int v = 0; v < 8; v++)
                    Oacc[dt][nt][v] *= sc;

            // Build P as a B operand [K=key x N=q] in registers:
            // lane<16 needs keys 0..15 ; lane>=16 needs keys 16..31 (2 bf16 per VGPR).
            unsigned pk0[4], pk1[4], o0[4], o1[4];
            #pragma unroll
            for (int j = 0; j < 4; j++) {
                pk0[j] = packbf(pv[2 * j],     pv[2 * j + 1]);       // own keys (acc0 block)
                pk1[j] = packbf(pv[8 + 2 * j], pv[8 + 2 * j + 1]);   // own keys (acc1 block)
            }
            #pragma unroll
            for (int j = 0; j < 4; j++) {
                o0[j] = (unsigned)__shfl_xor((int)pk0[j], 16, 32);
                o1[j] = (unsigned)__shfl_xor((int)pk1[j], 16, 32);
            }
            union { v16bf v; unsigned u[8]; } Bp;
            #pragma unroll
            for (int j = 0; j < 4; j++) {
                Bp.u[j]     = half ? o1[j]  : pk0[j];
                Bp.u[4 + j] = half ? pk1[j] : o0[j];
            }
            // O^T += V^T * P : A = Vt rows (4 d-tiles), keys = this chunk
            #pragma unroll
            for (int dt = 0; dt < 4; dt++) {
                v16bf va = load_tile_a(Vp + (size_t)(dt * 16) * SEQ + kc, SEQ, lane);
                Oacc[dt][nt] = wmma_bf(va, Bp.v, Oacc[dt][nt]);
            }
        }
    }

    // O^T C-layout: lane = q, vgprs = 8 consecutive d -> packed b128 row-major O stores
    #pragma unroll
    for (int nt = 0; nt < 2; nt++) {
        float inv = 1.f / lrun[nt];
        int q = q0 + nt * 16 + r;
        size_t rowbase = ((size_t)b * SEQ + q) * HID + head * DKV;
        #pragma unroll
        for (int dt = 0; dt < 4; dt++) {
            union { bf16_t h8[8]; v4u qv; } pk;
            #pragma unroll
            for (int v = 0; v < 8; v++) pk.h8[v] = f2bf(Oacc[dt][nt][v] * inv);
            *(v4u*)(Ob + rowbase + dt * 16 + half * 8) = pk.qv;
        }
    }
}

// --------------------------------------------------------- output projection
__global__ __launch_bounds__(256) void outproj_kernel(
        const bf16_t* __restrict__ Ob, const bf16_t* __restrict__ Wob,
        float* __restrict__ out) {
    int lane = threadIdx.x & 31;
    int wave = threadIdx.x >> 5;
    int wm = wave & 3, wn = wave >> 2;
    int M0 = blockIdx.x * 128 + wm * 32;
    int N0 = blockIdx.y * 128 + wn * 64;

    v8f acc[2][4] = {};
    for (int k0 = 0; k0 < HID; k0 += 32) {
        v16bf a0 = load_tile_a(Ob + (size_t)(M0 +  0) * HID + k0, HID, lane);
        v16bf a1 = load_tile_a(Ob + (size_t)(M0 + 16) * HID + k0, HID, lane);
        #pragma unroll
        for (int nt = 0; nt < 4; nt++) {
            v16bf bt = load_tile_b(Wob + (size_t)(N0 + nt * 16) * HID + k0, HID, lane);
            acc[0][nt] = wmma_bf(a0, bt, acc[0][nt]);
            acc[1][nt] = wmma_bf(a1, bt, acc[1][nt]);
        }
    }
    int r = lane & 15, mb = (lane >> 4) * 8;
    #pragma unroll
    for (int mt = 0; mt < 2; mt++)
        #pragma unroll
        for (int nt = 0; nt < 4; nt++) {
            int ng = N0 + nt * 16 + r;
            int mg = M0 + mt * 16 + mb;
            #pragma unroll
            for (int v = 0; v < 8; v++)
                out[(size_t)(mg + v) * HID + ng] = acc[mt][nt][v];
        }
}

// ---------------------------------------------------------------------- host
extern "C" void kernel_launch(void* const* d_in, const int* in_sizes, int n_in,
                              void* d_out, int out_size, void* d_ws, size_t ws_size,
                              hipStream_t stream) {
    const float* X    = (const float*)d_in[0];
    const float* mask = (const float*)d_in[1];
    const float* Wq   = (const float*)d_in[2];
    const float* Wk   = (const float*)d_in[3];
    const float* Wv   = (const float*)d_in[4];
    const float* Wo   = (const float*)d_in[5];

    const size_t NX = (size_t)MTOT * HID;                 // 3,145,728
    const size_t NW = (size_t)HID * HID;                  //   589,824
    const size_t NQ = (size_t)BATCH * NHEADS * SEQ * DKV; // == NX

    bf16_t* p   = (bf16_t*)d_ws;
    bf16_t* Xb  = p; p += NX;
    bf16_t* Wqb = p; p += NW;
    bf16_t* Wkb = p; p += NW;
    bf16_t* Wvb = p; p += NW;
    bf16_t* Wob = p; p += NW;
    bf16_t* Qb  = p; p += NQ;
    bf16_t* Kb  = p; p += NQ;
    bf16_t* Vtb = p; p += NQ;
    bf16_t* Ob  = p; p += NX;

    cvt_kernel<<<(int)((NX / 4 + 255) / 256), 256, 0, stream>>>(X,  Xb,  (int)NX);
    cvt_kernel<<<(int)((NW / 4 + 255) / 256), 256, 0, stream>>>(Wq, Wqb, (int)NW);
    cvt_kernel<<<(int)((NW / 4 + 255) / 256), 256, 0, stream>>>(Wk, Wkb, (int)NW);
    cvt_kernel<<<(int)((NW / 4 + 255) / 256), 256, 0, stream>>>(Wv, Wvb, (int)NW);
    cvt_kernel<<<(int)((NW / 4 + 255) / 256), 256, 0, stream>>>(Wo, Wob, (int)NW);

    qkv_kernel<<<dim3(MTOT / 128, HID / 128, 3), 256, 0, stream>>>(
        Xb, Wqb, Wkb, Wvb, Qb, Kb, Vtb);

    attn_kernel<<<dim3(SEQ / 256, BATCH * NHEADS), 256, 0, stream>>>(
        Qb, Kb, Vtb, mask, Ob);

    outproj_kernel<<<dim3(MTOT / 128, HID / 128), 256, 0, stream>>>(
        Ob, Wob, (float*)d_out);
}
